// MambaBlock_57354993271128
// MI455X (gfx1250) — matrix operand order repaired and checked by reading, not compiled, not run on hardware
//
#include <hip/hip_runtime.h>
#include <hip/hip_bf16.h>
#include <math.h>

// Mamba block (NL=2) for MI455X / gfx1250, wave32.
// GEMMs: bf16-resident operands, async global->LDS double-buffered staging
// (GLOBAL_LOAD_ASYNC_TO_LDS_B128 / ASYNCcnt), v_wmma_f32_16x16x32_bf16 with
// f32 accumulate. Norms / conv / dt-softplus / selective scan stay f32.

#define NL 2
#define D_MODEL 512
#define D_INNER 2048
#define D_STATE 16
#define D_CONV 4
#define DT_RANK 32
#define BSZ 4
#define SEQ 1024
#define TOK (BSZ * SEQ)   // 4096 token rows

typedef __attribute__((ext_vector_type(16))) __bf16 v16bf;
typedef __attribute__((ext_vector_type(8)))  float  v8f;

union BFrag { v16bf v; __bf16 e[16]; uint4 q[2]; };

// ---- CDNA5 async global->LDS copy (16B per lane, tracked by ASYNCcnt) ----
__device__ __forceinline__ void async_ld_b128(unsigned lds_off, const void* gaddr) {
  asm volatile("global_load_async_to_lds_b128 %0, %1, off"
               :: "v"(lds_off), "v"(gaddr) : "memory");
}
template <int N>
__device__ __forceinline__ void wait_asynccnt() {
  asm volatile("s_wait_asynccnt %0" :: "i"(N) : "memory");
}
__device__ __forceinline__ unsigned lds_off_of(const void* p) {
  // generic shared address = {shared_aperture, lds_offset[31:0]}
  return (unsigned)(uintptr_t)p;
}

// ---------------------------------------------------------------------------
// C[M,N] f32 = A[M,K] bf16 @ W[N,K]^T bf16
//
// Block: 256 threads = 8 waves; block tile 256(M) x 64(N); wave tile 32x64
// (2 A-frags x 4 B-frags = 8 WMMAs per K=32 step). A (256x32) and B (64x32)
// bf16 tiles are DMA'd into double-buffered LDS with async loads: issue tile
// t+1, s_wait_asynccnt 5 (async loads complete in order => tile t resident),
// barrier, compute. Requires M%256==0, N%64==0, K%32==0 (holds for
// 4096x4096x512, 4096x64x2048, 4096x512x2048).
//
// Fragment layouts per CDNA5 ISA 7.12.2 (wave32), lane = g*16 + r:
//   A 16x32 bf16: lane holds row r; elems 0..7 -> K = g*8.., 8..15 -> 16+g*8..
//   B 32x16 bf16: lane holds col r; elems 0..7 -> K = g*8.., 8..15 -> 16+g*8..
//   C 16x16 f32 : lane holds col r; vgpr i -> row = i + 8g
// ---------------------------------------------------------------------------
__global__ __launch_bounds__(256)
void gemm_wmma_bf16(const __bf16* __restrict__ A, const __bf16* __restrict__ W,
                    float* __restrict__ C, int M, int N, int K) {
  __shared__ __align__(16) __bf16 sA[2][256 * 32];   // 2 x 16 KB
  __shared__ __align__(16) __bf16 sB[2][64 * 32];    // 2 x  4 KB

  const int tid   = threadIdx.x;
  const int lane  = tid & 31;
  const int wave  = tid >> 5;
  const int g     = lane >> 4;
  const int r     = lane & 15;
  const int blkM  = blockIdx.y * 256;
  const int tileN = blockIdx.x * 64;

  v8f acc[2][4];
#pragma unroll
  for (int mi = 0; mi < 2; ++mi)
#pragma unroll
    for (int j = 0; j < 4; ++j) acc[mi][j] = (v8f){};

  // stage one K=32 tile into LDS buffer `buf` via 5 async B128s per thread
  auto stage = [&](int buf, int k0) {
#pragma unroll
    for (int i = 0; i < 4; ++i) {                    // A: 1024 16B chunks
      const int c   = i * 256 + tid;
      const int row = c >> 2;
      const int seg = (c & 3) << 3;                  // elem offset, 16B chunks
      async_ld_b128(lds_off_of(&sA[buf][row * 32 + seg]),
                    A + (size_t)(blkM + row) * K + k0 + seg);
    }
    {                                                // B: 256 16B chunks
      const int row = tid >> 2;
      const int seg = (tid & 3) << 3;
      async_ld_b128(lds_off_of(&sB[buf][row * 32 + seg]),
                    W + (size_t)(tileN + row) * K + k0 + seg);
    }
  };

  const int ntiles = K >> 5;
  stage(0, 0);
  for (int t = 0; t < ntiles; ++t) {
    const int cur = t & 1;
    if (t + 1 < ntiles) {
      stage(1 - cur, (t + 1) << 5);
      wait_asynccnt<5>();      // tile t's 5 loads retired (in-order)
    } else {
      wait_asynccnt<0>();
    }
    __syncthreads();

    BFrag a0, a1;
    {
      const int row0 = wave * 32 + r;
      const int row1 = row0 + 16;
      a0.q[0] = *(const uint4*)&sA[cur][row0 * 32 + g * 8];
      a0.q[1] = *(const uint4*)&sA[cur][row0 * 32 + 16 + g * 8];
      a1.q[0] = *(const uint4*)&sA[cur][row1 * 32 + g * 8];
      a1.q[1] = *(const uint4*)&sA[cur][row1 * 32 + 16 + g * 8];
    }
#pragma unroll
    for (int j = 0; j < 4; ++j) {
      BFrag b;
      const int nrow = j * 16 + r;
      b.q[0] = *(const uint4*)&sB[cur][nrow * 32 + g * 8];
      b.q[1] = *(const uint4*)&sB[cur][nrow * 32 + 16 + g * 8];
      acc[0][j] = __builtin_amdgcn_wmma_f32_16x16x32_bf16(
          false, a0.v, false, b.v, (short)0, acc[0][j], false, false);
      acc[1][j] = __builtin_amdgcn_wmma_f32_16x16x32_bf16(
          false, a1.v, false, b.v, (short)0, acc[1][j], false, false);
    }
    __syncthreads();
  }

  // ---- store 32x64 per wave ----
#pragma unroll
  for (int mi = 0; mi < 2; ++mi) {
#pragma unroll
    for (int j = 0; j < 4; ++j) {
      const int col = tileN + j * 16 + r;
#pragma unroll
      for (int i = 0; i < 8; ++i) {
        const int row = blkM + wave * 32 + mi * 16 + i + g * 8;
        C[(size_t)row * N + col] = acc[mi][j][i];
      }
    }
  }
}

// ---------------------------------------------------------------------------
// One-shot f32 -> bf16 conversion (weights; runs once per launch).
// ---------------------------------------------------------------------------
__global__ __launch_bounds__(256)
void cvt_bf16_kernel(const float* __restrict__ src, __bf16* __restrict__ dst) {
  const int i = (blockIdx.x * 256 + threadIdx.x) * 4;
  const float4 v = *(const float4*)(src + i);
  __bf16* p = dst + i;
  p[0] = (__bf16)v.x; p[1] = (__bf16)v.y; p[2] = (__bf16)v.z; p[3] = (__bf16)v.w;
}

// ---------------------------------------------------------------------------
// Fused residual add + RMSNorm over D_MODEL=512.
//   r = h (+ res_in); res_out = r (optional); y = r*rsqrt(mean(r^2)+eps)*w
//   y written as f32 (outf) and/or bf16 (outb) — bf16 feeds the WMMA GEMM.
// ---------------------------------------------------------------------------
__global__ __launch_bounds__(256)
void addnorm_kernel(const float* __restrict__ h, const float* __restrict__ res_in,
                    float* __restrict__ res_out, const float* __restrict__ w,
                    float* __restrict__ outf, __bf16* __restrict__ outb) {
  __shared__ float red[256];
  const int t = threadIdx.x;
  const size_t base = (size_t)blockIdx.x * D_MODEL;

  float r0 = h[base + t];
  float r1 = h[base + t + 256];
  if (res_in) { r0 += res_in[base + t]; r1 += res_in[base + t + 256]; }
  if (res_out) { res_out[base + t] = r0; res_out[base + t + 256] = r1; }

  red[t] = r0 * r0 + r1 * r1;
  __syncthreads();
#pragma unroll
  for (int s = 128; s > 0; s >>= 1) {
    if (t < s) red[t] += red[t + s];
    __syncthreads();
  }
  const float scale = rsqrtf(red[0] * (1.0f / D_MODEL) + 1e-5f);
  const float y0 = r0 * scale * w[t];
  const float y1 = r1 * scale * w[t + 256];
  if (outf) { outf[base + t] = y0; outf[base + t + 256] = y1; }
  if (outb) { outb[base + t] = (__bf16)y0; outb[base + t + 256] = (__bf16)y1; }
}

// ---------------------------------------------------------------------------
// Causal depthwise conv1d (kernel 4) + bias + SiLU.
// x lives in xz[:, :, 0:D_INNER] (row stride 2*D_INNER).
// Writes f32 (for the scan's x input) and bf16 (for the x_proj GEMM).
// ---------------------------------------------------------------------------
__global__ __launch_bounds__(256)
void conv_silu_kernel(const float* __restrict__ xz, const float* __restrict__ cw,
                      const float* __restrict__ cb, float* __restrict__ xcf,
                      __bf16* __restrict__ xcb) {
  const int idx = blockIdx.x * 256 + threadIdx.x;   // b*L*Din + l*Din + d
  const int d = idx & (D_INNER - 1);
  const int l = (idx >> 11) & (SEQ - 1);
  const int b = idx >> 21;

  float s = cb[d];
#pragma unroll
  for (int k = 0; k < D_CONV; ++k) {
    const int lp = l + k - (D_CONV - 1);
    if (lp >= 0)
      s = fmaf(xz[(size_t)(b * SEQ + lp) * (2 * D_INNER) + d], cw[d * D_CONV + k], s);
  }
  const float v = s / (1.0f + __expf(-s));          // silu
  xcf[idx] = v;
  xcb[idx] = (__bf16)v;
}

// ---------------------------------------------------------------------------
// Fused selective scan (dt_proj + softplus fused; dtw row cached in VGPRs):
//   dt = softplus(dbl[:, :32] . dtw[d] + dtb[d])
//   h[n] = h[n]*exp(dt*A[d,n]) + dt*x*B[n];  y = <h,C> + x*D[d]
//   out = y * silu(z), written as bf16 (feeds out_proj GEMM directly).
// One thread per (b,d); the 64-float (dt_in|B|C) vector per (b,t) staged in LDS.
// ---------------------------------------------------------------------------
__global__ __launch_bounds__(256)
void scan_kernel(const float* __restrict__ dbl, const float* __restrict__ xcf,
                 const float* __restrict__ xz, const float* __restrict__ dtw,
                 const float* __restrict__ dtb, const float* __restrict__ A_log,
                 const float* __restrict__ Dp, __bf16* __restrict__ yout) {
  __shared__ float sBC[DT_RANK + 2 * D_STATE];   // 64 floats
  const int b = blockIdx.y;
  const int d = blockIdx.x * 256 + threadIdx.x;

  float Arow[D_STATE];
#pragma unroll
  for (int n = 0; n < D_STATE; ++n) Arow[n] = -__expf(A_log[d * D_STATE + n]);
  float wdt[DT_RANK];
#pragma unroll
  for (int j = 0; j < DT_RANK; ++j) wdt[j] = dtw[d * DT_RANK + j];
  const float bias = dtb[d];
  const float Dd   = Dp[d];

  float st[D_STATE];
#pragma unroll
  for (int n = 0; n < D_STATE; ++n) st[n] = 0.0f;

  for (int t = 0; t < SEQ; ++t) {
    __syncthreads();
    if (threadIdx.x < 64)
      sBC[threadIdx.x] = dbl[(size_t)(b * SEQ + t) * 64 + threadIdx.x];
    __syncthreads();

    float dtv = bias;
#pragma unroll
    for (int j = 0; j < DT_RANK; ++j) dtv = fmaf(wdt[j], sBC[j], dtv);
    const float dt = (dtv > 20.0f) ? dtv : log1pf(__expf(dtv));

    const size_t tok = (size_t)(b * SEQ + t);
    const float xv = xcf[tok * D_INNER + d];
    const float z  = xz[tok * (2 * D_INNER) + D_INNER + d];
    const float dx = dt * xv;

    float y = 0.0f;
#pragma unroll
    for (int n = 0; n < D_STATE; ++n) {
      const float dA = __expf(dt * Arow[n]);
      st[n] = fmaf(st[n], dA, dx * sBC[DT_RANK + n]);
      y = fmaf(st[n], sBC[DT_RANK + D_STATE + n], y);
    }
    y = fmaf(xv, Dd, y);
    const float sig = 1.0f / (1.0f + __expf(-z));
    yout[tok * D_INNER + d] = (__bf16)(y * (z * sig));
  }
}

// ---------------------------------------------------------------------------
extern "C" void kernel_launch(void* const* d_in, const int* in_sizes, int n_in,
                              void* d_out, int out_size, void* d_ws, size_t ws_size,
                              hipStream_t stream) {
  (void)in_sizes; (void)n_in; (void)out_size; (void)ws_size;

  const float* x      = (const float*)d_in[0];
  const float* norm_w = (const float*)d_in[1];
  const float* in_w   = (const float*)d_in[2];
  const float* conv_w = (const float*)d_in[3];
  const float* conv_b = (const float*)d_in[4];
  const float* xp_w   = (const float*)d_in[5];
  const float* dt_w   = (const float*)d_in[6];
  const float* dt_b   = (const float*)d_in[7];
  const float* A_log  = (const float*)d_in[8];
  const float* Dp     = (const float*)d_in[9];
  const float* out_w  = (const float*)d_in[10];
  const float* fin_w  = (const float*)d_in[11];

  // ---- workspace carve-up ----
  float* ws   = (float*)d_ws;
  float* resB = ws;                                 // 4096*512 f32
  float* hB   = resB + (size_t)TOK * D_MODEL;       // 4096*512 f32
  float* xzB  = hB   + (size_t)TOK * D_MODEL;       // 4096*4096 f32
  float* xcB  = xzB  + (size_t)TOK * 2 * D_INNER;   // 4096*2048 f32 (conv out)
  float* dblB = xcB  + (size_t)TOK * D_INNER;       // 4096*64 f32
  __bf16* bw  = (__bf16*)(dblB + (size_t)TOK * 64);
  __bf16* hnBf   = bw;                                      // 4096*512
  __bf16* xcBf   = hnBf  + (size_t)TOK * D_MODEL;           // 4096*2048
  __bf16* inWbf  = xcBf  + (size_t)TOK * D_INNER;           // 2*4096*512
  __bf16* xpWbf  = inWbf + (size_t)NL * 2 * D_INNER * D_MODEL;  // 2*64*2048
  __bf16* outWbf = xpWbf + (size_t)NL * (DT_RANK + 2 * D_STATE) * D_INNER; // 2*512*2048

  // ---- one-shot weight conversion to bf16 (element counts % 1024 == 0) ----
  cvt_bf16_kernel<<<(NL * 2 * D_INNER * D_MODEL) / 1024, 256, 0, stream>>>(in_w, inWbf);
  cvt_bf16_kernel<<<(NL * (DT_RANK + 2 * D_STATE) * D_INNER) / 1024, 256, 0, stream>>>(xp_w, xpWbf);
  cvt_bf16_kernel<<<(NL * D_MODEL * D_INNER) / 1024, 256, 0, stream>>>(out_w, outWbf);

  const float* hcur = x;
  const float* rin  = nullptr;

  for (int i = 0; i < NL; ++i) {
    // prenorm: res = h (+res); hn = rmsnorm(res)*w  (bf16 for WMMA)
    addnorm_kernel<<<TOK, 256, 0, stream>>>(hcur, rin, resB, norm_w + i * D_MODEL,
                                            nullptr, hnBf);
    // in_proj: (4096,512) @ (4096,512)^T -> xz (4096,4096)
    gemm_wmma_bf16<<<dim3(2 * D_INNER / 64, TOK / 256), 256, 0, stream>>>(
        hnBf, inWbf + (size_t)i * 2 * D_INNER * D_MODEL, xzB, TOK, 2 * D_INNER, D_MODEL);
    // depthwise causal conv + silu on x half (f32 for scan, bf16 for x_proj)
    conv_silu_kernel<<<(TOK * D_INNER) / 256, 256, 0, stream>>>(
        xzB, conv_w + (size_t)i * D_INNER * D_CONV, conv_b + (size_t)i * D_INNER,
        xcB, xcBf);
    // x_proj: (4096,2048) @ (64,2048)^T -> dbl (4096,64)
    gemm_wmma_bf16<<<dim3(64 / 64, TOK / 256), 256, 0, stream>>>(
        xcBf, xpWbf + (size_t)i * (DT_RANK + 2 * D_STATE) * D_INNER, dblB,
        TOK, DT_RANK + 2 * D_STATE, D_INNER);
    // fused dt_proj + softplus + selective scan + D-skip + silu(z) gating;
    // gated y overwrites xcBf (x_proj already consumed it)
    scan_kernel<<<dim3(D_INNER / 256, BSZ), 256, 0, stream>>>(
        dblB, xcB, xzB, dt_w + (size_t)i * D_INNER * DT_RANK, dt_b + (size_t)i * D_INNER,
        A_log + (size_t)i * D_INNER * D_STATE, Dp + (size_t)i * D_INNER, xcBf);
    // out_proj: (4096,2048) @ (512,2048)^T -> h (4096,512)
    gemm_wmma_bf16<<<dim3(D_MODEL / 64, TOK / 256), 256, 0, stream>>>(
        xcBf, outWbf + (size_t)i * D_MODEL * D_INNER, hB, TOK, D_MODEL, D_INNER);
    hcur = hB;
    rin  = resB;
  }
  // final: rmsnorm(h + res) * final_w -> out (f32)
  addnorm_kernel<<<TOK, 256, 0, stream>>>(hB, resB, nullptr, fin_w,
                                          (float*)d_out, nullptr);
}
